// EcfpModel_56977036149216
// MI455X (gfx1250) — compile-verified, matrix operand order checked by reading, not run on hardware
//
#include <hip/hip_runtime.h>
#include <hip/hip_bf16.h>
#include <cstdint>

// Problem sizes (fixed by the reference)
#define NB   2048   // batch
#define KD   2048   // ecfp_len
#define FPD  512    // fp_len
#define NM1  2047   // NB-1
#define EPSV 1e-5f

typedef __attribute__((ext_vector_type(16))) _Float16 v16h;
typedef __attribute__((ext_vector_type(8)))  float    v8f;
typedef __attribute__((ext_vector_type(4)))  int      v4i_t;

// ---- CDNA5 async global->LDS support (guarded; sync fallback) -------------
#if defined(__has_builtin)
#  if __has_builtin(__builtin_amdgcn_global_load_async_to_lds_b128)
#    define USE_ASYNC_LDS 1
#  endif
#endif

__device__ __forceinline__ void copy16_to_lds(float* lds, const float* g) {
#ifdef USE_ASYNC_LDS
    // signature (probe-verified): (v4i AS1* gsrc, v4i AS3* ldst, imm off, imm cpol)
    __builtin_amdgcn_global_load_async_to_lds_b128(
        (__attribute__((address_space(1))) v4i_t*)const_cast<float*>(g),
        (__attribute__((address_space(3))) v4i_t*)lds,
        /*offset=*/0, /*cpol=*/0);
#else
    const float4 t = *(const float4*)g;
    *(float4*)lds = t;
#endif
}

__device__ __forceinline__ void wait_async_lds() {
#ifdef USE_ASYNC_LDS
#  if __has_builtin(__builtin_amdgcn_s_wait_asynccnt)
    __builtin_amdgcn_s_wait_asynccnt(0);
#  else
    asm volatile("s_wait_asynccnt 0" ::: "memory");
#  endif
#endif
}

// ---------------------------------------------------------------------------
// Kernel 1: fps = ecfps @ W^T + b  via V_WMMA_F32_16X16X32_F16
// One wave computes a 16(M) x 32(N) tile: two WMMAs per K-step sharing the A
// fragment. Tiles: 128 (M) x 16 (N-pairs) = 2048 waves = 256 blocks * 8 waves.
// ---------------------------------------------------------------------------
__global__ __launch_bounds__(256) void gemm_wmma_kernel(
    const float* __restrict__ ecfps,   // [NB][KD]
    const float* __restrict__ Wm,      // [FPD][KD]
    const float* __restrict__ bias,    // [FPD]
    float* __restrict__ fps)           // [NB][FPD]
{
    const int lane = threadIdx.x & 31;
    const int wave = threadIdx.x >> 5;
    const int tile = blockIdx.x * 8 + wave;          // 0..2047
    const int tileN = tile & 15;                     // 16 col-pair tiles
    const int tileM = tile >> 4;                     // 128 row tiles

    const int r    = lane & 15;                      // A-row / B-col owned by lane
    const int half = lane >> 4;                      // K-half selector

    const int m_base = tileM * 16;
    const int n_base = tileN * 32;

    const float* __restrict__ arow  = ecfps + (size_t)(m_base + r) * KD;
    const float* __restrict__ brow0 = Wm + (size_t)(n_base + r) * KD;        // B = W^T
    const float* __restrict__ brow1 = Wm + (size_t)(n_base + 16 + r) * KD;

    v8f c0 = {};
    v8f c1 = {};

    for (int k0 = 0; k0 < KD; k0 += 32) {
        const int kb = k0 + half * 8;
        float ta[16], tb0[16], tb1[16];
        #pragma unroll
        for (int e = 0; e < 8; ++e) {
            ta[e]      = arow[kb + e];
            ta[e + 8]  = arow[kb + 16 + e];
            tb0[e]     = brow0[kb + e];
            tb0[e + 8] = brow0[kb + 16 + e];
            tb1[e]     = brow1[kb + e];
            tb1[e + 8] = brow1[kb + 16 + e];
        }
        v16h A, B0, B1;
        #pragma unroll
        for (int e = 0; e < 16; ++e) {
            A[e]  = (_Float16)ta[e];
            B0[e] = (_Float16)tb0[e];
            B1[e] = (_Float16)tb1[e];
        }
        c0 = __builtin_amdgcn_wmma_f32_16x16x32_f16(
                false, A, false, B0, (short)0, c0, false, false);
        c1 = __builtin_amdgcn_wmma_f32_16x16x32_f16(
                false, A, false, B1, (short)0, c1, false, false);
    }

    // C/D layout: VGPR v, lane L -> M = v + 8*(L>=16), N = L%16
    const int n0 = n_base + (lane & 15);
    const int n1 = n0 + 16;
    const float bv0 = bias[n0];
    const float bv1 = bias[n1];
    #pragma unroll
    for (int v = 0; v < 8; ++v) {
        const int mIdx = m_base + v + 8 * half;
        fps[(size_t)mIdx * FPD + n0] = c0[v] + bv0;
        fps[(size_t)mIdx * FPD + n1] = c1[v] + bv1;
    }
}

// ---------------------------------------------------------------------------
// Kernel 2: full pairwise L1 distance d[i][j] = sum_k |fps[i,k]-fps[j,k]|
// 64x64 pair tile / block, K in chunks of 64, double-buffered async->LDS,
// float4 LDS reads (ds_load_b128) to keep the loop VALU-bound.
// ---------------------------------------------------------------------------
#define TI 64
#define TJ 64
#define TK 64
#define TKP 68            // padded row stride: 272 B (16B aligned, conflict-free)
#define NCHUNK (FPD / TK) // 8

__global__ __launch_bounds__(256) void l1_dist_kernel(
    const float* __restrict__ fps,     // [NB][FPD]
    float* __restrict__ d)             // [NB][NB]
{
    __shared__ __align__(16) float As[2][TI][TKP];
    __shared__ __align__(16) float Bs[2][TJ][TKP];

    const int bi = blockIdx.x >> 5;    // 32 tiles along i
    const int bj = blockIdx.x & 31;    // 32 tiles along j
    const int i0 = bi * TI;
    const int j0 = bj * TJ;

    const int tid = threadIdx.x;
    const int tx = tid & 15;           // j sub-lane
    const int ty = tid >> 4;           // i sub-lane

    // per-thread segment addresses for cooperative B128 copies:
    // 1024 x 16B segments per 64x64 tile -> 4 segments/thread per matrix
    int seg_row[4], seg_col[4];
    #pragma unroll
    for (int s = 0; s < 4; ++s) {
        const int lin = s * 256 + tid;     // b128 unit index, 16/row
        seg_row[s] = lin >> 4;
        seg_col[s] = (lin & 15) * 4;
    }

    float acc[4][4];
    #pragma unroll
    for (int u = 0; u < 4; ++u)
        #pragma unroll
        for (int v = 0; v < 4; ++v) acc[u][v] = 0.0f;

    // prologue: stage chunk 0 into buffer 0
    #pragma unroll
    for (int s = 0; s < 4; ++s) {
        copy16_to_lds(&As[0][seg_row[s]][seg_col[s]],
                      fps + (size_t)(i0 + seg_row[s]) * FPD + seg_col[s]);
        copy16_to_lds(&Bs[0][seg_row[s]][seg_col[s]],
                      fps + (size_t)(j0 + seg_row[s]) * FPD + seg_col[s]);
    }
    wait_async_lds();
    __syncthreads();

    for (int c = 0; c < NCHUNK; ++c) {
        const int cur = c & 1;

        // issue next chunk into the alternate buffer
        if (c + 1 < NCHUNK) {
            const int k1 = (c + 1) * TK;
            #pragma unroll
            for (int s = 0; s < 4; ++s) {
                copy16_to_lds(&As[cur ^ 1][seg_row[s]][seg_col[s]],
                              fps + (size_t)(i0 + seg_row[s]) * FPD + k1 + seg_col[s]);
                copy16_to_lds(&Bs[cur ^ 1][seg_row[s]][seg_col[s]],
                              fps + (size_t)(j0 + seg_row[s]) * FPD + k1 + seg_col[s]);
            }
        }

        // compute on current buffer: float4 LDS reads, 16B-aligned
        for (int kk = 0; kk < TK; kk += 4) {
            float4 a4[4], b4[4];
            #pragma unroll
            for (int u = 0; u < 4; ++u)
                a4[u] = *(const float4*)&As[cur][ty + 16 * u][kk];
            #pragma unroll
            for (int v = 0; v < 4; ++v)
                b4[v] = *(const float4*)&Bs[cur][tx + 16 * v][kk];
            #pragma unroll
            for (int u = 0; u < 4; ++u)
                #pragma unroll
                for (int v = 0; v < 4; ++v) {
                    acc[u][v] += fabsf(a4[u].x - b4[v].x);
                    acc[u][v] += fabsf(a4[u].y - b4[v].y);
                    acc[u][v] += fabsf(a4[u].z - b4[v].z);
                    acc[u][v] += fabsf(a4[u].w - b4[v].w);
                }
        }

        wait_async_lds();
        __syncthreads();
    }

    #pragma unroll
    for (int u = 0; u < 4; ++u) {
        const int i = i0 + ty + 16 * u;
        #pragma unroll
        for (int v = 0; v < 4; ++v) {
            const int j = j0 + tx + 16 * v;
            d[(size_t)i * NB + j] = acc[u][v];
        }
    }
}

// ---------------------------------------------------------------------------
// Kernel 3a: coalesced partial BatchNorm stats.
// Grid = 8 column-groups x 32 row-slices. Thread t owns column jp = cg*256+t
// and accumulates over 64 rows (consecutive threads -> consecutive j:
// fully coalesced). Fixed-order accumulation => deterministic.
// ---------------------------------------------------------------------------
__global__ __launch_bounds__(256) void bn_partial_kernel(
    const float* __restrict__ d,         // [NB][NB]
    float* __restrict__ colsum_part,     // [32][2048]
    float* __restrict__ colsq_part)      // [32][2048]
{
    const int cg = blockIdx.x & 7;       // column group
    const int rs = blockIdx.x >> 3;      // row slice (0..31)
    const int jp = cg * 256 + threadIdx.x;
    if (jp >= NM1) return;

    float s = 0.0f, sq = 0.0f;
    const int iend = rs * 64 + 64;
    for (int i = rs * 64; i < iend; ++i) {
        const int j = jp + (jp >= i ? 1 : 0);
        const float v = d[(size_t)i * NB + j];
        s += v;
        sq += v * v;
    }
    colsum_part[rs * 2048 + jp] = s;
    colsq_part[rs * 2048 + jp]  = sq;
}

// ---------------------------------------------------------------------------
// Kernel 3b: combine partials -> mean / invstd per column.
// ---------------------------------------------------------------------------
__global__ __launch_bounds__(256) void bn_combine_kernel(
    const float* __restrict__ colsum_part,  // [32][2048]
    const float* __restrict__ colsq_part,   // [32][2048]
    float* __restrict__ mean,               // [NM1]
    float* __restrict__ invstd)             // [NM1]
{
    const int jp = blockIdx.x * 256 + threadIdx.x;
    if (jp >= NM1) return;

    float s = 0.0f, sq = 0.0f;
    #pragma unroll
    for (int r = 0; r < 32; ++r) {
        s  += colsum_part[r * 2048 + jp];
        sq += colsq_part[r * 2048 + jp];
    }
    const float m = s / (float)NB;
    const float var = sq / (float)NB - m * m;   // biased (jnp.var default)
    mean[jp]   = m;
    invstd[jp] = rsqrtf(var + EPSV);
}

// ---------------------------------------------------------------------------
// Kernel 4: BCE-with-logits partial sums, one block per row i.
// labels[i] scalar; d-row, mean, invstd, labels[j] reads all coalesced.
// ---------------------------------------------------------------------------
__global__ __launch_bounds__(256) void loss_kernel(
    const float* __restrict__ d,       // [NB][NB]
    const float* __restrict__ mean,    // [NM1]
    const float* __restrict__ invstd,  // [NM1]
    const int*   __restrict__ labels,  // [NB]
    float* __restrict__ partials)      // [NB]
{
    const int i   = blockIdx.x;        // row
    const int tid = threadIdx.x;
    const float li = (float)labels[i];
    const float* drow = d + (size_t)i * NB;

    float acc = 0.0f;
    for (int jp = tid; jp < NM1; jp += 256) {
        const int j = jp + (jp >= i ? 1 : 0);
        const float dn = (drow[j] - mean[jp]) * invstd[jp];
        const float tgt = 1.0f - li * (float)labels[j];
        // stable softplus(dn) - dn*tgt
        const float sp = fmaxf(dn, 0.0f) + log1pf(expf(-fabsf(dn)));
        acc += sp - dn * tgt;
    }

    __shared__ float sred[256];
    sred[tid] = acc;
    __syncthreads();
    for (int off = 128; off > 0; off >>= 1) {
        if (tid < off) sred[tid] += sred[tid + off];
        __syncthreads();
    }
    if (tid == 0) partials[i] = sred[0];
}

// ---------------------------------------------------------------------------
// Kernel 5: final reduction -> scalar mean
// ---------------------------------------------------------------------------
__global__ __launch_bounds__(256) void final_reduce_kernel(
    const float* __restrict__ partials,   // [NB]
    float* __restrict__ out)
{
    const int tid = threadIdx.x;
    float acc = 0.0f;
    for (int i = tid; i < NB; i += 256) acc += partials[i];

    __shared__ float sred[256];
    sred[tid] = acc;
    __syncthreads();
    for (int off = 128; off > 0; off >>= 1) {
        if (tid < off) sred[tid] += sred[tid + off];
        __syncthreads();
    }
    if (tid == 0) out[0] = sred[0] / ((float)NB * (float)NM1);
}

// ---------------------------------------------------------------------------
// Host-side launcher
// ---------------------------------------------------------------------------
extern "C" void kernel_launch(void* const* d_in, const int* in_sizes, int n_in,
                              void* d_out, int out_size, void* d_ws, size_t ws_size,
                              hipStream_t stream) {
    const float* ecfps  = (const float*)d_in[0];   // [2048*2048]
    const int*   labels = (const int*)d_in[1];     // [2048*1]
    const float* Wm     = (const float*)d_in[2];   // [512*2048]
    const float* bias   = (const float*)d_in[3];   // [512]

    // Workspace carve-up (all 256B aligned). Total ~20.6 MB.
    uint8_t* ws = (uint8_t*)d_ws;
    const size_t MB = 1024 * 1024;
    float* fps         = (float*)(ws);                        //  4 MB
    float* dmat        = (float*)(ws + 4 * MB);               // 16 MB
    float* mean        = (float*)(ws + 20 * MB);              //  8 KB
    float* invstd      = (float*)(ws + 20 * MB + 16384);      //  8 KB
    float* partials    = (float*)(ws + 20 * MB + 32768);      //  8 KB
    float* colsum_part = (float*)(ws + 20 * MB + 65536);      // 256 KB
    float* colsq_part  = (float*)(ws + 20 * MB + 65536 + 262144);

    // 1) fps = ecfps @ W^T + b  (WMMA f16 -> f32, 2 tiles/wave)
    gemm_wmma_kernel<<<256, 256, 0, stream>>>(ecfps, Wm, bias, fps);

    // 2) d[i][j] = L1(fps[i], fps[j])  (async double-buffered LDS tiles)
    l1_dist_kernel<<<32 * 32, 256, 0, stream>>>(fps, dmat);

    // 3) per-column (post remove_diag) mean / invstd, coalesced two-stage
    bn_partial_kernel<<<8 * 32, 256, 0, stream>>>(dmat, colsum_part, colsq_part);
    bn_combine_kernel<<<8, 256, 0, stream>>>(colsum_part, colsq_part, mean, invstd);

    // 4) partial BCE sums (one block per row)
    loss_kernel<<<NB, 256, 0, stream>>>(dmat, mean, invstd, labels, partials);

    // 5) scalar mean
    final_reduce_kernel<<<1, 256, 0, stream>>>(partials, (float*)d_out);
}
